// DeepSeekMoEGate_12627203850988
// MI455X (gfx1250) — compile-verified, matrix operand order checked by reading, not run on hardware
//
#include <hip/hip_runtime.h>
#include <hip/hip_bf16.h>

typedef __attribute__((ext_vector_type(16))) __bf16 v16bf;
typedef __attribute__((ext_vector_type(8)))  __bf16 v8bf;   // 16 bytes
typedef __attribute__((ext_vector_type(8)))  float  v8f;
typedef __attribute__((ext_vector_type(4)))  float  v4f;
typedef __attribute__((ext_vector_type(4)))  int    v4i;    // 16 bytes

#define H_DIM 7168
#define E_DIM 256
#define NT 16                 // 256 experts / 16-wide n-tiles
#define KC_N (H_DIM / 32)     // 224 K-steps
#define T_TOK (4 * 4096)

#if __has_builtin(__builtin_amdgcn_global_load_async_to_lds_b128)
#define HAS_ASYNC_LDS 1
#else
#define HAS_ASYNC_LDS 0
#endif

// pointer-to-AS-qualified-v4i types expected by the async builtin
typedef __attribute__((address_space(1))) v4i* gv4ip;   // global
typedef __attribute__((address_space(3))) v4i* lv4ip;   // LDS

__device__ __forceinline__ void wait_asynccnt0() {
#if __has_builtin(__builtin_amdgcn_s_wait_asynccnt)
  __builtin_amdgcn_s_wait_asynccnt(0);
#else
  asm volatile("s_wait_asynccnt 0x0" ::: "memory");
#endif
}

// ---------------------------------------------------------------------------
// Prep: convert W [E,H] f32 -> bf16, swizzled into WMMA-B fragment order.
// Fragment f = kc*16 + n (1024 B each): [reg r(0|1)][lane][8 bf16].
// Lane l of fragment (kc,n) holds expert e = n*16 + (l&15),
// K = kc*32 + (l>>4)*8 + r*16 + {0..7}.  One thread emits one 16 B chunk.
// ---------------------------------------------------------------------------
__global__ __launch_bounds__(256) void moe_w_swizzle(
    const float* __restrict__ w, v8bf* __restrict__ wbf)
{
  const int o    = blockIdx.x * 256 + threadIdx.x;  // 16B chunk id
  const int f    = o >> 6;                          // 64 chunks / fragment
  const int q    = o & 63;
  const int r    = q >> 5;
  const int lane = q & 31;
  const int kc   = f >> 4;
  const int n    = f & 15;
  const int e    = n * 16 + (lane & 15);
  const int kb   = kc * 32 + (lane >> 4) * 8 + r * 16;

  const float* src = w + (size_t)e * H_DIM + kb;
  v4f s0 = *(const v4f*)(src);
  v4f s1 = *(const v4f*)(src + 4);
  v8bf p;
  #pragma unroll
  for (int j = 0; j < 4; ++j) {
    p[j]     = (__bf16)s0[j];
    p[4 + j] = (__bf16)s1[j];
  }
  wbf[o] = p;
}

// ---------------------------------------------------------------------------
// GEMM: logits[T,E] = x[T,H] @ W[E,H]^T  via v_wmma_f32_16x16x32_bf16.
// Block = 8 waves = 128 tokens; each wave owns 16 tokens x 256 experts.
// B (swizzled bf16) staged once per block through double-buffered LDS using
// GLOBAL_LOAD_ASYNC_TO_LDS_B128 (ASYNCcnt); x streamed once from HBM (NT).
// B fragments 4-way software-pipelined in registers so ds_load latency
// overlaps the matrix pipe instead of a dscnt==0 stall per WMMA.
// ---------------------------------------------------------------------------
__global__ __launch_bounds__(256) void moe_gate_gemm(
    const float* __restrict__ x,      // [T, H]
    const v4i* __restrict__ wbf,      // swizzled bf16 W, 16KB per kc
    float* __restrict__ logits)       // [T, E]
{
  __shared__ v4i Bbuf[2][1024];       // 2 x 16 KB

  const int lane  = threadIdx.x;          // 0..31
  const int wave  = threadIdx.y;          // 0..7
  const int tid   = wave * 32 + lane;     // 0..255
  const int mtile = blockIdx.x * 8 + wave;
  const int half  = lane >> 4;
  const int l16   = lane & 15;

  const float* arow = x + (size_t)(mtile * 16 + l16) * H_DIM + half * 8;

  // ---- preload B chunk 0 ----
#if HAS_ASYNC_LDS
  {
    const v4i* g0 = wbf + tid;
    v4i*       l0 = &Bbuf[0][tid];
    #pragma unroll
    for (int j = 0; j < 4; ++j)
      __builtin_amdgcn_global_load_async_to_lds_b128(
          (gv4ip)(g0 + j * 256), (lv4ip)(l0 + j * 256), 0, 0);
    wait_asynccnt0();
  }
#else
  #pragma unroll
  for (int j = 0; j < 4; ++j) Bbuf[0][tid + j * 256] = wbf[tid + j * 256];
#endif
  __syncthreads();

  v8f acc[NT];
  #pragma unroll
  for (int n = 0; n < NT; ++n) acc[n] = (v8f){0.f,0.f,0.f,0.f,0.f,0.f,0.f,0.f};

  struct P { v8bf lo; v8bf hi; };
  union  U { P p; v16bf v; };

  for (int kc = 0; kc < KC_N; ++kc) {
    const int  cur  = kc & 1;
    const bool more = (kc + 1 < KC_N);

#if HAS_ASYNC_LDS
    // kick off async global->LDS staging of the next B chunk (no VGPR pass)
    if (more) {
      const v4i* gn = wbf + (size_t)(kc + 1) * 1024 + tid;
      v4i*       ln = &Bbuf[cur ^ 1][tid];
      #pragma unroll
      for (int j = 0; j < 4; ++j)
        __builtin_amdgcn_global_load_async_to_lds_b128(
            (gv4ip)(gn + j * 256), (lv4ip)(ln + j * 256), 0, 0);
    }
#else
    v4i stage[4];
    if (more) {
      const v4i* src = wbf + (size_t)(kc + 1) * 1024;
      #pragma unroll
      for (int j = 0; j < 4; ++j) stage[j] = src[tid + j * 256];
    }
#endif

    // A fragment: K = kc*32 + half*8 + {0..7, 16..23}
    const float* ap = arow + kc * 32;
    v4f a0 = __builtin_nontemporal_load((const v4f*)(ap));
    v4f a1 = __builtin_nontemporal_load((const v4f*)(ap + 4));
    v4f a2 = __builtin_nontemporal_load((const v4f*)(ap + 16));
    v4f a3 = __builtin_nontemporal_load((const v4f*)(ap + 20));
    v16bf a;
    #pragma unroll
    for (int j = 0; j < 4; ++j) {
      a[j]      = (__bf16)a0[j];
      a[4 + j]  = (__bf16)a1[j];
      a[8 + j]  = (__bf16)a2[j];
      a[12 + j] = (__bf16)a3[j];
    }

    // sweep 16 expert tiles in groups of 4: issue all 8 ds_loads of the
    // group (4 live fragment buffers -> distinct registers), then 4 WMMAs.
    // In-order DS returns => staggered waits (dscnt<=6/4/2/0), so only the
    // first WMMA of a group can see LDS latency.
    const v8bf* bbase = (const v8bf*)&Bbuf[cur][0];
    #pragma unroll
    for (int n4 = 0; n4 < NT; n4 += 4) {
      U u0, u1, u2, u3;
      u0.p.lo = bbase[(n4 + 0) * 64 + lane];
      u0.p.hi = bbase[(n4 + 0) * 64 + 32 + lane];
      u1.p.lo = bbase[(n4 + 1) * 64 + lane];
      u1.p.hi = bbase[(n4 + 1) * 64 + 32 + lane];
      u2.p.lo = bbase[(n4 + 2) * 64 + lane];
      u2.p.hi = bbase[(n4 + 2) * 64 + 32 + lane];
      u3.p.lo = bbase[(n4 + 3) * 64 + lane];
      u3.p.hi = bbase[(n4 + 3) * 64 + 32 + lane];
      acc[n4 + 0] = __builtin_amdgcn_wmma_f32_16x16x32_bf16(
          false, a, false, u0.v, (short)0, acc[n4 + 0], false, false);
      acc[n4 + 1] = __builtin_amdgcn_wmma_f32_16x16x32_bf16(
          false, a, false, u1.v, (short)0, acc[n4 + 1], false, false);
      acc[n4 + 2] = __builtin_amdgcn_wmma_f32_16x16x32_bf16(
          false, a, false, u2.v, (short)0, acc[n4 + 2], false, false);
      acc[n4 + 3] = __builtin_amdgcn_wmma_f32_16x16x32_bf16(
          false, a, false, u3.v, (short)0, acc[n4 + 3], false, false);
    }

#if HAS_ASYNC_LDS
    if (more) wait_asynccnt0();       // own async stores to LDS complete
#else
    if (more) {
      #pragma unroll
      for (int j = 0; j < 4; ++j) Bbuf[cur ^ 1][tid + j * 256] = stage[j];
    }
#endif
    __syncthreads();                  // everyone's staging visible
  }

  // store: C/D layout is M = v + 8*half, N = l16
  #pragma unroll
  for (int n = 0; n < NT; ++n) {
    #pragma unroll
    for (int v = 0; v < 8; ++v) {
      const int m = mtile * 16 + half * 8 + v;
      logits[(size_t)m * E_DIM + n * 16 + l16] = acc[n][v];
    }
  }
}

// ---------------------------------------------------------------------------
// Gating: one wave per token. Lane l owns experts [8l, 8l+8).
// Group g (32 experts) == lane cluster [4g, 4g+4).
// ---------------------------------------------------------------------------
__global__ __launch_bounds__(256) void moe_gate_topk(
    const float* __restrict__ logits,   // [T, E]
    const float* __restrict__ bias,     // [E]
    float* __restrict__ out_w,          // [T, 8]
    int* __restrict__ out_i)            // [T, 8]
{
  const int lane = threadIdx.x;
  const int wave = threadIdx.y;
  const int t    = blockIdx.x * 8 + wave;

  const float* lp = logits + (size_t)t * E_DIM + lane * 8;
  float s[8];
  #pragma unroll
  for (int j = 0; j < 8; ++j) {
    float z   = lp[j];
    float sig = 1.0f / (1.0f + __expf(-z));
    s[j] = sig + bias[lane * 8 + j];
  }

  // group max: local over 8, then over the 4-lane cluster
  float lm = s[0];
  #pragma unroll
  for (int j = 1; j < 8; ++j) lm = fmaxf(lm, s[j]);
  lm = fmaxf(lm, __shfl_xor(lm, 1, 32));
  lm = fmaxf(lm, __shfl_xor(lm, 2, 32));

  // broadcast 8 group maxima, rank my group (lower index wins ties)
  float gm[8];
  #pragma unroll
  for (int g = 0; g < 8; ++g) gm[g] = __shfl(lm, g * 4, 32);
  const int myg = lane >> 2;
  int rank = 0;
  #pragma unroll
  for (int g = 0; g < 8; ++g)
    rank += (gm[g] > gm[myg]) || ((gm[g] == gm[myg]) && (g < myg));
  const bool sel = (rank < 4);            // topk_group = 4
  #pragma unroll
  for (int j = 0; j < 8; ++j) s[j] = sel ? s[j] : 0.0f;

  // top-8 over 256: 8 rounds of wave-wide argmax (lower index on ties)
  float wsum  = 0.0f;
  float wkeep = 0.0f;
  int   ikeep = 0;
  #pragma unroll
  for (int it = 0; it < 8; ++it) {
    float bm = s[0];
    int   bi = lane * 8;
    #pragma unroll
    for (int j = 1; j < 8; ++j)
      if (s[j] > bm) { bm = s[j]; bi = lane * 8 + j; }
    #pragma unroll
    for (int d = 1; d < 32; d <<= 1) {
      float om = __shfl_xor(bm, d, 32);
      int   oi = __shfl_xor(bi, d, 32);
      if (om > bm || (om == bm && oi < bi)) { bm = om; bi = oi; }
    }
    if (lane == it) { wkeep = bm; ikeep = bi; }
    wsum += bm;
    const int local = bi - lane * 8;
    #pragma unroll
    for (int j = 0; j < 8; ++j)
      if (j == local) s[j] = -1.0f;
  }

  const float inv = 1.0f / (wsum + 1e-6f);
  if (lane < 8) {
    out_w[(size_t)t * 8 + lane] = wkeep * inv;
    out_i[(size_t)t * 8 + lane] = ikeep;
  }
}

extern "C" void kernel_launch(void* const* d_in, const int* in_sizes, int n_in,
                              void* d_out, int out_size, void* d_ws, size_t ws_size,
                              hipStream_t stream) {
  const float* x    = (const float*)d_in[0];   // [4,4096,7168]
  const float* w    = (const float*)d_in[1];   // [256,7168]
  const float* bias = (const float*)d_in[2];   // [256]
  // d_in[3..5] = n_group=8, topk_group=4, top_k=8 (hardcoded)

  float* logits = (float*)d_ws;                                   // 16 MB
  char*  wbf    = (char*)d_ws + (size_t)T_TOK * E_DIM * 4;        // +3.67 MB

  dim3 blk(32, 8);                              // 8 wave32 waves / block
  // swizzle W: E*H/8 = 229376 chunks of 16B -> 896 blocks of 256 threads
  moe_w_swizzle<<<dim3(896), dim3(256), 0, stream>>>(w, (v8bf*)wbf);
  moe_gate_gemm<<<dim3(T_TOK / 16 / 8), blk, 0, stream>>>(x, (const v4i*)wbf, logits);

  float* out_w = (float*)d_out;                     // [T,8] weights
  int*   out_i = (int*)d_out + (size_t)T_TOK * 8;   // [T,8] indices
  moe_gate_topk<<<dim3(T_TOK / 8), blk, 0, stream>>>(logits, bias, out_w, out_i);
}
